// VectorQuantizer_446676599464
// MI455X (gfx1250) — matrix operand order repaired and checked by reading, not compiled, not run on hardware
//
#include <hip/hip_runtime.h>
#include <hip/hip_bf16.h>

typedef __attribute__((ext_vector_type(16))) _Float16 v16h;
typedef __attribute__((ext_vector_type(8)))  float    v8f;
typedef __attribute__((ext_vector_type(2)))  _Float16 h2;
typedef __attribute__((ext_vector_type(4)))  unsigned int v4u;
typedef __attribute__((ext_vector_type(8)))  int      v8i;
typedef __attribute__((ext_vector_type(4)))  int      v4i;

#define NB     16
#define SLEN   1024
#define DDIM   64
#define NCODE  8192
#define ROWS   (NB*SLEN)      // 16384
#define NTILES (NCODE/16)     // 512
#define MTILES (ROWS/16)      // 1024

#define CHUNK_TILES 16                    // N-tiles per TDM chunk
#define CHUNK_ROWS  (CHUNK_TILES*16)      // 256 embedding rows
#define CHUNK_ELEMS (CHUNK_ROWS*DDIM)     // 16384 halfs = 32 KB
#define NCHUNK      (NTILES/CHUNK_TILES)  // 32
#define LROW        72                    // LDS row stride in halfs (128B + 16B pad)

union U16H { uint4 q[2]; v16h h; };

// ---------------- Phase 0a: normalize inputs, emit f16 x + scale ----------------
__global__ void k_prep_x(const float* __restrict__ in, _Float16* __restrict__ xh,
                         float* __restrict__ scale) {
    const int wave = threadIdx.x >> 5, lane = threadIdx.x & 31;
    const int row  = blockIdx.x * 8 + wave;
    float2 iv = ((const float2*)in)[row * 32 + lane];
    float ss = iv.x * iv.x + iv.y * iv.y;
    #pragma unroll
    for (int m = 16; m > 0; m >>= 1) ss += __shfl_xor(ss, m, 32);
    float nrm = sqrtf(ss);
    if (lane == 0) scale[row] = nrm;
    h2 hv; hv[0] = (_Float16)(iv.x / nrm); hv[1] = (_Float16)(iv.y / nrm);
    ((h2*)xh)[row * 32 + lane] = hv;
}

// ---------------- Phase 0b: embedding -> f16 + row squared norms ----------------
__global__ void k_prep_e(const float* __restrict__ emb, _Float16* __restrict__ eh,
                         float* __restrict__ esq) {
    const int wave = threadIdx.x >> 5, lane = threadIdx.x & 31;
    const int row  = blockIdx.x * 8 + wave;
    float2 ev = ((const float2*)emb)[row * 32 + lane];
    float ss = ev.x * ev.x + ev.y * ev.y;
    #pragma unroll
    for (int m = 16; m > 0; m >>= 1) ss += __shfl_xor(ss, m, 32);
    if (lane == 0) esq[row] = ss;
    h2 hv; hv[0] = (_Float16)ev.x; hv[1] = (_Float16)ev.y;
    ((h2*)eh)[row * 32 + lane] = hv;
}

__global__ void k_zero(int* __restrict__ cnts) {
    int i = blockIdx.x * 256 + threadIdx.x;
    if (i < NCODE) cnts[i] = 0;
}

// --- TDM: DMA one 32KB embedding chunk (16 tiles) into LDS with 16B/row padding ---
// D#: 1-D tile of 16384 x 2B elems; pad 4 DWORDs every 32 DWORDs -> LDS row
// stride 144B so ds_load_b128 of 16 rows hits 16 distinct bank groups.
__device__ __forceinline__ void tdm_load_chunk(const _Float16* gsrc, unsigned lds_off) {
    unsigned long long ga = (unsigned long long)(uintptr_t)gsrc;
    v4u g0 = { 1u,                                   // count=1 (valid descriptor)
               lds_off,                              // lds_addr [63:32]
               (unsigned)ga,                         // global_addr[31:0]
               (unsigned)(ga >> 32) | 0x80000000u }; // global_addr[56:32] | type=2
    v8i g1 = { (int)((1u << 16)      // data_size = 1 (2 bytes)
                   | (1u << 20)      // pad_enable
                   | (4u << 22)      // pad_interval: 32 DWORDs (128B)
                   | (3u << 25)),    // pad_amount: 4 DWORDs (16B)
               (int)(16384u << 16),  // tensor_dim0[15:0]=16384 (atomic_addr=0)
               (int)(1u << 16),      // tensor_dim0[31:16]=0, tensor_dim1=1
               (int)(16384u << 16),  // tile_dim0 = 16384 elems
               1,                    // tile_dim1 = 1, tile_dim2 = 0
               16384,                // tensor_dim0_stride
               0, 0 };
    v4i z4 = { 0, 0, 0, 0 };
    v8i z8 = { 0, 0, 0, 0, 0, 0, 0, 0 };
    __builtin_amdgcn_tensor_load_to_lds(g0, g1, z4, z4, z8, 0);
}

// ------------- Phase 1: WMMA distance GEMM + fused per-row min/argmin -------------
// 8 waves share double-buffered LDS B-tiles (TDM copies overlap WMMA compute);
// global embedding traffic drops 8x vs streaming per-wave.
__global__ void __launch_bounds__(256)
k_dist_argmin(const _Float16* __restrict__ xh, const _Float16* __restrict__ eh,
              const float* __restrict__ esq,
              float* __restrict__ mind, int* __restrict__ amin) {
    __shared__ __align__(16) _Float16 lds[2][CHUNK_ROWS * LROW]; // 2 x 36,864 B

    const int wave = threadIdx.x >> 5, lane = threadIdx.x & 31;
    const int mt = blockIdx.x * 8 + wave;
    const int lh = lane & 15;

    // A-matrix (16-bit 16x32 layout): lane = row M=lh;
    // lanes 0-15 hold K {0..7,16..23}, lanes 16-31 hold K {8..15,24..31}
    const uint4* qa = (const uint4*)(xh + (size_t)(mt * 16 + lh) * DDIM);
    const int sa = (lane < 16) ? 0 : 1;
    U16H A0, A1;
    A0.q[0] = qa[sa];     A0.q[1] = qa[sa + 2];
    A1.q[0] = qa[sa + 4]; A1.q[1] = qa[sa + 6];

    // B-matrix (16-bit 32x16 layout): lane = column N=lh;
    // lanes 0-15 hold K {0..7,16..23}, lanes 16-31 hold K {8..15,24..31}
    const int sb = (lane < 16) ? 0 : 1;

    float minv[8]; int mini[8];
    #pragma unroll
    for (int r = 0; r < 8; ++r) { minv[r] = 3.4e38f; mini[r] = 0; }

    if (wave == 0)
        tdm_load_chunk(eh, (unsigned)(uintptr_t)&lds[0][0]);

    for (int c = 0; c < NCHUNK; ++c) {
        if (wave == 0) __builtin_amdgcn_s_wait_tensorcnt(0);  // chunk c landed
        __syncthreads();                                      // publish to all waves
        if (wave == 0 && c + 1 < NCHUNK)                      // prefetch next chunk
            tdm_load_chunk(eh + (size_t)(c + 1) * CHUNK_ELEMS,
                           (unsigned)(uintptr_t)&lds[(c + 1) & 1][0]);

        const _Float16* buf = &lds[c & 1][0];
        #pragma unroll 4
        for (int t = 0; t < CHUNK_TILES; ++t) {
            const uint4* qb = (const uint4*)(buf + (size_t)(t * 16 + lh) * LROW);
            U16H B0, B1;
            B0.q[0] = qb[sb];     B0.q[1] = qb[sb + 2];   // K = 0..31
            B1.q[0] = qb[sb + 4]; B1.q[1] = qb[sb + 6];   // K = 32..63

            v8f cacc = {};
            cacc = __builtin_amdgcn_wmma_f32_16x16x32_f16(false, A0.h, false, B0.h,
                                                          (short)0, cacc, false, false);
            cacc = __builtin_amdgcn_wmma_f32_16x16x32_f16(false, A1.h, false, B1.h,
                                                          (short)0, cacc, false, false);

            const int j = c * (CHUNK_TILES * 16) + t * 16 + lh;
            const float ej = esq[j];
            #pragma unroll
            for (int r = 0; r < 8; ++r) {
                float s = fmaf(-2.0f, cacc[r], ej);        // ||e||^2 - 2 x.e
                if (s < minv[r]) { minv[r] = s; mini[r] = j; }
            }
        }
    }

    // reduce across the 16 lanes of each half; tie-break to smallest index
    for (int m = 1; m <= 8; m <<= 1) {
        #pragma unroll
        for (int r = 0; r < 8; ++r) {
            float om = __shfl_xor(minv[r], m, 32);
            int   oi = __shfl_xor(mini[r], m, 32);
            if (om < minv[r] || (om == minv[r] && oi < mini[r])) {
                minv[r] = om; mini[r] = oi;
            }
        }
    }
    if (lane == 0) {
        #pragma unroll
        for (int r = 0; r < 8; ++r) {
            mind[mt * 16 + r] = minv[r] + 1.0f;            // + ||x||^2 (== 1)
            amin[mt * 16 + r] = mini[r];
        }
    }
    if (lane == 16) {
        #pragma unroll
        for (int r = 0; r < 8; ++r) {
            mind[mt * 16 + 8 + r] = minv[r] + 1.0f;
            amin[mt * 16 + 8 + r] = mini[r];
        }
    }
}

// ------- Phase 2: per-batch bitonic argsort of distances + faithful gather -------
__global__ void k_sort(const float* __restrict__ mind, const int* __restrict__ amin,
                       int* __restrict__ encf, float* __restrict__ outenc) {
    __shared__ float sd[SLEN];
    __shared__ int   si[SLEN];
    const int t = threadIdx.x, b = blockIdx.x;
    sd[t] = mind[b * SLEN + t];
    si[t] = t;
    __syncthreads();
    for (int k = 2; k <= SLEN; k <<= 1) {
        for (int j = k >> 1; j > 0; j >>= 1) {
            int x = t ^ j;
            if (x > t) {
                float da = sd[t], db = sd[x];
                int   ia = si[t], ib = si[x];
                bool agtb = (da > db) || (da == db && ia > ib); // lexicographic => stable
                bool up   = ((t & k) == 0);
                if (agtb == up) { sd[t] = db; sd[x] = da; si[t] = ib; si[x] = ia; }
            }
            __syncthreads();
        }
    }
    int ord = si[t];          // in [0,1024)
    int e   = amin[ord];      // flat gather -> batch 0's argmins (reference quirk)
    encf[b * SLEN + t]   = e;
    outenc[b * SLEN + t] = (float)e;
}

// ---------- Phase 3: quantized output, per-row sq-err, code histogram ----------
__global__ void k_quant(const float* __restrict__ in, const float* __restrict__ emb,
                        const float* __restrict__ scale, const int* __restrict__ encf,
                        float* __restrict__ rsum, int* __restrict__ cnts,
                        float* __restrict__ outq) {
    const int wave = threadIdx.x >> 5, lane = threadIdx.x & 31;
    const int row  = blockIdx.x * 8 + wave;
    const int e    = encf[row];
    const float sc = scale[row];
    float2 ev = ((const float2*)emb)[e * 32 + lane];
    float2 iv = ((const float2*)in)[row * 32 + lane];
    float x0 = iv.x / sc, x1 = iv.y / sc;
    float d0 = ev.x - x0, d1 = ev.y - x1;
    float acc = d0 * d0 + d1 * d1;
    float* o = outq + (size_t)row * DDIM + lane * 2;   // d_out+1: 4B aligned only
    o[0] = ev.x * sc;
    o[1] = ev.y * sc;
    #pragma unroll
    for (int m = 16; m > 0; m >>= 1) acc += __shfl_xor(acc, m, 32);
    if (lane == 0) {
        rsum[row] = acc;
        atomicAdd(&cnts[e], 1);                        // integer: deterministic
    }
}

// --------------- Phase 4: deterministic loss + perplexity reduction ---------------
__global__ void k_final(const float* __restrict__ rsum, const int* __restrict__ cnts,
                        float* __restrict__ out) {
    __shared__ float s1[1024];
    __shared__ float s2[1024];
    const int t = threadIdx.x;
    float a = 0.0f;
    for (int i = t; i < ROWS; i += 1024) a += rsum[i];
    float ent = 0.0f;
    for (int i = t; i < NCODE; i += 1024) {
        float p = (float)cnts[i] * (1.0f / (float)ROWS);
        ent += p * logf(p + 1e-10f);
    }
    s1[t] = a; s2[t] = ent;
    __syncthreads();
    for (int s = 512; s > 0; s >>= 1) {
        if (t < s) { s1[t] += s1[t + s]; s2[t] += s2[t + s]; }
        __syncthreads();
    }
    if (t == 0) {
        out[0] = 1.99f * (s1[0] / (float)(ROWS * DDIM));   // q + 0.99*e latent loss
        out[1 + ROWS * DDIM] = expf(-s2[0]);               // perplexity
    }
}

extern "C" void kernel_launch(void* const* d_in, const int* in_sizes, int n_in,
                              void* d_out, int out_size, void* d_ws, size_t ws_size,
                              hipStream_t stream) {
    (void)in_sizes; (void)n_in; (void)out_size; (void)ws_size;
    const float* in  = (const float*)d_in[0];   // [16,1024,64] f32
    const float* emb = (const float*)d_in[1];   // [8192,64]    f32
    float* out = (float*)d_out;                 // [loss | quant 1048576 | perp | enc 16384]

    uint8_t* w = (uint8_t*)d_ws;
    _Float16* xh   = (_Float16*)(w);                         // 2,097,152 B
    _Float16* eh   = (_Float16*)(w + 2097152);               // 1,048,576 B
    float*    scl  = (float*)  (w + 3145728);                //    65,536 B
    float*    esq  = (float*)  (w + 3211264);                //    32,768 B
    float*    mind = (float*)  (w + 3244032);                //    65,536 B
    int*      amin = (int*)    (w + 3309568);                //    65,536 B
    int*      encf = (int*)    (w + 3375104);                //    65,536 B
    int*      cnts = (int*)    (w + 3440640);                //    32,768 B
    float*    rsum = (float*)  (w + 3473408);                //    65,536 B

    k_zero<<<NCODE / 256, 256, 0, stream>>>(cnts);
    k_prep_x<<<ROWS / 8, 256, 0, stream>>>(in, xh, scl);
    k_prep_e<<<NCODE / 8, 256, 0, stream>>>(emb, eh, esq);
    k_dist_argmin<<<MTILES / 8, 256, 0, stream>>>(xh, eh, esq, mind, amin);
    k_sort<<<NB, SLEN, 0, stream>>>(mind, amin, encf, out + 1 + ROWS * DDIM + 1);
    k_quant<<<ROWS / 8, 256, 0, stream>>>(in, emb, scl, encf, rsum, cnts, out + 1);
    k_final<<<1, 1024, 0, stream>>>(rsum, cnts, out);
}
